// MACoreAttention_81999515615567
// MI455X (gfx1250) — matrix operand order repaired and checked by reading, not compiled, not run on hardware
//
#include <hip/hip_runtime.h>
#include <math.h>

// CDNA5 WMMA vector types
typedef __attribute__((ext_vector_type(16))) _Float16 v16h;
typedef __attribute__((ext_vector_type(8)))  float    v8f;

#define HEADS 8
#define HDIM  64
#define WIN   64
#define TILE  16

union PK { unsigned u; _Float16 h[2]; };

// Convert 8 contiguous f32 (16B aligned) into halfs [base, base+8) of a v16h.
__device__ __forceinline__ void cvt8(v16h& r, int base, const float* p) {
  const float4* p4 = (const float4*)p;
  float4 a = p4[0], b = p4[1];
  r[base + 0] = (_Float16)a.x; r[base + 1] = (_Float16)a.y;
  r[base + 2] = (_Float16)a.z; r[base + 3] = (_Float16)a.w;
  r[base + 4] = (_Float16)b.x; r[base + 5] = (_Float16)b.y;
  r[base + 6] = (_Float16)b.z; r[base + 7] = (_Float16)b.w;
}

__device__ __forceinline__ v16h cvt16(const float* p) {
  v16h r;
  cvt8(r, 0, p);
  cvt8(r, 8, p + 8);
  return r;
}

// Stage the 32x64 f32 V tile [kb, kb+32) x [0,64) into LDS with coalesced
// async b128 copies (GLOBAL_LOAD_ASYNC_TO_LDS_B128, tracked by ASYNCcnt).
// 512 x 16B chunks, 16 per lane. LDS operand = low 32 bits of flat address
// (LDS aperture, ISA 10.2).
__device__ __forceinline__ void stage_v_async(const float* __restrict__ V,
                                              int kb, int h, int S,
                                              float* ldsVbuf, int lane) {
#pragma unroll
  for (int t = 0; t < 16; ++t) {
    const int chunk = lane + 32 * t;   // 0..511
    const int row   = chunk >> 4;      // key row in tile, 0..31
    const int c16   = chunk & 15;      // 16B chunk within the 256B row
    int kr = kb + row;
    if (kr >= S) kr = S - 1;
    const float* gp = V + ((size_t)kr * HEADS + h) * HDIM + c16 * 4;
    const unsigned           lds_off = (unsigned)(uintptr_t)(ldsVbuf + row * HDIM + c16 * 4);
    const unsigned long long gaddr   = (unsigned long long)(uintptr_t)gp;
    asm volatile("global_load_async_to_lds_b128 %0, %1, off"
                 :: "v"(lds_off), "v"(gaddr)
                 : "memory");
  }
}

// One wave32 per (query-tile, head); flash-attention over the +/-64 window.
// 32 keys per step; V tiles double-buffered through LDS via async copies.
__global__ __launch_bounds__(32)
void swattn_wmma_kernel(const float* __restrict__ Q,
                        const float* __restrict__ K,
                        const float* __restrict__ V,
                        float* __restrict__ O,
                        int S) {
  // P transpose bounce: word [row][j] packs f16 probs for keys j and j+16.
  __shared__ unsigned ldsP[TILE][TILE];
  // Double-buffered raw f32 V tiles (2 x 8 KB).
  __shared__ float ldsV[2][2 * TILE * HDIM];

  const int lane = threadIdx.x;   // 0..31 (wave32)
  const int grp  = lane >> 4;     // half-wave group
  const int l16  = lane & 15;
  const int qt   = blockIdx.x;
  const int h    = blockIdx.y;
  const int q0   = qt * TILE;
  const float scale = 0.125f;     // 1/sqrt(64), applied post-WMMA

  // ---- Q A-fragments (16x32 f16 each; two cover D=64).
  // A layout: lane row M = l16; halfs i -> K=16g+i.
  const float* qrow = Q + ((size_t)(q0 + l16) * HEADS + h) * HDIM;
  v16h aq0, aq1;
  cvt8(aq0, 0, qrow +       16 * grp);
  cvt8(aq0, 8, qrow + 8 +   16 * grp);
  cvt8(aq1, 0, qrow + 32 +  16 * grp);
  cvt8(aq1, 8, qrow + 40 +  16 * grp);

  // Online-softmax state; C-layout element r <-> query row (r + 8*grp).
  float m[8], l[8];
  v8f   o[4];
#pragma unroll
  for (int r = 0; r < 8; ++r) { m[r] = -1e30f; l[r] = 0.0f; }
#pragma unroll
  for (int c = 0; c < 4; ++c) { v8f z = {}; o[c] = z; }

  const int ntiles  = S / TILE;
  const int kt0     = (qt - 4 > 0) ? (qt - 4) : 0;
  const int kt1     = (qt + 4 < ntiles - 1) ? (qt + 4) : (ntiles - 1);
  const int kb0     = kt0 * TILE;
  const int kb_last = kt1 * TILE;

  // Prologue: kick off async staging of the first V tile.
  stage_v_async(V, kb0, h, S, ldsV[0], lane);

  int step = 0;
  for (int kb = kb0; kb <= kb_last; kb += 2 * TILE, ++step) {
    const int  buf      = step & 1;
    const bool has_next = (kb + 2 * TILE <= kb_last);

    // Overlap: stage the next V tile while we do QK + softmax on this one.
    if (has_next)
      stage_v_async(V, kb + 2 * TILE, h, S, ldsV[buf ^ 1], lane);

    // Prefetch the K rows of the next 32-key step (global_prefetch_b8).
    if (has_next) {
      const float* np = K + ((size_t)(kb + 2 * TILE + l16) * HEADS + h) * HDIM;
      __builtin_prefetch(np, 0, 1);
    }

    // ---- K B-fragments for the two 16-key tiles.
    // B layout: lane column N = l16; halfs j: d = 16g+j / 32+16g+j.
    const int key0 = kb + l16;
    const int key1 = kb + TILE + l16;
    const int k0c  = (key0 < S) ? key0 : (S - 1);
    const int k1c  = (key1 < S) ? key1 : (S - 1);
    const float* kp0 = K + ((size_t)k0c * HEADS + h) * HDIM;
    const float* kp1 = K + ((size_t)k1c * HEADS + h) * HDIM;
    v16h b00 = cvt16(kp0 +      16 * grp);
    v16h b01 = cvt16(kp0 + 32 + 16 * grp);
    v16h b10 = cvt16(kp1 +      16 * grp);
    v16h b11 = cvt16(kp1 + 32 + 16 * grp);

    // ---- S = Q.K^T for 32 keys: two 16x16 f32 tiles, 4 WMMAs.
    v8f s0 = {}, s1 = {};
    s0 = __builtin_amdgcn_wmma_f32_16x16x32_f16(false, aq0, false, b00,
                                                (short)0, s0, false, false);
    s0 = __builtin_amdgcn_wmma_f32_16x16x32_f16(false, aq1, false, b01,
                                                (short)0, s0, false, false);
    s1 = __builtin_amdgcn_wmma_f32_16x16x32_f16(false, aq0, false, b10,
                                                (short)0, s1, false, false);
    s1 = __builtin_amdgcn_wmma_f32_16x16x32_f16(false, aq1, false, b11,
                                                (short)0, s1, false, false);

    // ---- mask + online softmax over 32 keys (one reduction pass).
#pragma unroll
    for (int r = 0; r < 8; ++r) {
      const int qpos = q0 + r + 8 * grp;
      float sv0 = s0[r] * scale;
      float sv1 = s1[r] * scale;
      const bool v0 = (key0 < S) && (key0 >= qpos - WIN) && (key0 <= qpos + WIN);
      const bool v1 = (key1 < S) && (key1 >= qpos - WIN) && (key1 <= qpos + WIN);
      sv0 = v0 ? sv0 : -1e30f;
      sv1 = v1 ? sv1 : -1e30f;

      float rm = fmaxf(sv0, sv1);
#pragma unroll
      for (int off = 8; off >= 1; off >>= 1)
        rm = fmaxf(rm, __shfl_xor(rm, off, 32));

      const float mnew = fmaxf(m[r], rm);
      const float p0   = __expf(sv0 - mnew);
      const float p1   = __expf(sv1 - mnew);
      float rs = p0 + p1;
#pragma unroll
      for (int off = 8; off >= 1; off >>= 1)
        rs += __shfl_xor(rs, off, 32);

      const float alpha = __expf(m[r] - mnew);  // underflows to 0 on first step
      l[r] = l[r] * alpha + rs;
      m[r] = mnew;
#pragma unroll
      for (int c = 0; c < 4; ++c) o[c][r] *= alpha;

      // pack probs for keys (l16) and (l16+16) into one LDS word
      PK pk;
      pk.h[0] = (_Float16)p0;
      pk.h[1] = (_Float16)p1;
      ldsP[r + 8 * grp][l16] = pk.u;
    }
    __syncthreads();   // single wave: orders the LDS RAW (dscnt wait)

    // ---- P as A-fragment (16 rows x 32 keys), one b32 word per key pair.
    v16h ap;
#pragma unroll
    for (int i = 0; i < 8; ++i) {
      PK rd;
      rd.u = ldsP[l16][8 * grp + i];
      ap[i]     = rd.h[0];
      ap[8 + i] = rd.h[1];
    }

    // ---- Wait for THIS tile's async V staging; ASYNCcnt completes in order,
    // so <=16 leaves only the next tile's copies in flight.
    if (has_next)
      asm volatile("s_wait_asynccnt 16" ::: "memory");
    else
      asm volatile("s_wait_asynccnt 0" ::: "memory");

    // ---- O += P.V from the staged LDS tile. B layout: lane column = d
    // (16c + l16), halfs j<8 = key row 8g+j, j>=8 = key row 16+8g+(j-8). Conflict-free LDS gather
    // (consecutive lanes -> consecutive banks).
    const float* vt = ldsV[buf] + (8 * grp) * HDIM;
#pragma unroll
    for (int c = 0; c < 4; ++c) {
      v16h bv;
#pragma unroll
      for (int j = 0; j < 16; ++j)
        bv[j] = (_Float16)vt[(j < 8 ? j : j + 8) * HDIM + 16 * c + l16];
      o[c] = __builtin_amdgcn_wmma_f32_16x16x32_f16(false, ap, false, bv,
                                                    (short)0, o[c], false, false);
    }
    __syncthreads();   // protect ldsP/ldsV before the next step
  }

  // ---- finalize: divide by row sums, write out (C layout -> row-major).
#pragma unroll
  for (int r = 0; r < 8; ++r) {
    const float inv = 1.0f / l[r];
    const size_t row = (size_t)(q0 + r + 8 * grp) * HEADS + h;
#pragma unroll
    for (int c = 0; c < 4; ++c)
      O[row * HDIM + 16 * c + l16] = o[c][r] * inv;
  }
}

extern "C" void kernel_launch(void* const* d_in, const int* in_sizes, int n_in,
                              void* d_out, int out_size, void* d_ws, size_t ws_size,
                              hipStream_t stream) {
  const float* Q = (const float*)d_in[0];
  const float* K = (const float*)d_in[1];
  const float* V = (const float*)d_in[2];
  float*       O = (float*)d_out;

  const int S = in_sizes[0] / (HEADS * HDIM);   // B == 1
  dim3 grid(S / TILE, HEADS);
  swattn_wmma_kernel<<<grid, 32, 0, stream>>>(Q, K, V, O, S);
}
